// FlashMultiHeadAttention_49873160241204
// MI455X (gfx1250) — matrix operand order repaired and checked by reading, not compiled, not run on hardware
//
#include <hip/hip_runtime.h>
#include <hip/hip_bf16.h>

// ---------------------------------------------------------------------------
// GQA attention block for MI455X (gfx1250), wave32, WMMA f16 (f32 accum).
// Pipeline: Wq/Wk/Wv GEMMs (fp32 in -> fp32 out) -> RoPE+f16 pack -> flash
// attention (WMMA scores + online softmax + WMMA P*V) -> Wo GEMM -> fp32 out.
// Round 2: union-punned fragment loads (loads land directly in WMMA source
// regs), double-buffered LDS in the GEMM (1 barrier / K-step), global
// prefetch hints (global_prefetch_b8), b128 attention epilogue stores.
// ---------------------------------------------------------------------------

typedef _Float16 half_t;
typedef __attribute__((ext_vector_type(16))) _Float16 v16h;
typedef __attribute__((ext_vector_type(8)))  _Float16 v8h;
typedef __attribute__((ext_vector_type(8)))  float    v8f;

union FragH {  // 16 halfs = 8 VGPRs; h[0]/h[1] are the low/high b128 halves
  v16h v;
  v8h  h[2];
};

#define BB 4
#define CC 1024
#define TT 2048
#define HQ 16
#define HKV 4
#define DD 64
#define KVDIM 256

// LDS stride (halfs) for the GEMM X tile stored [n][k]: 48*2 = 96B, 16B-aligned
#define LDS_XK 48

// ---------------------------------------------------------------------------
// GEMM: Y[b][m][n] = sum_k W[m][k] * X[b][k][n] + bias[m]
// block = 256 threads (8 waves); block tile = 128(M) x 64(N); wave tile = 16x64.
// Double-buffered LDS staging of the X tile.
// ---------------------------------------------------------------------------
template <typename TX>
__global__ __launch_bounds__(256) void gemm_bias_wmma(
    const float* __restrict__ W,     // [M, K] row-major
    const TX*    __restrict__ X,     // [B, K, T]
    const float* __restrict__ bias,  // [M]
    float*       __restrict__ Y,     // [B, M, T]
    int M, int K)
{
  __shared__ half_t lds_x[2][64 * LDS_XK];

  const int b    = blockIdx.z;
  const int n0   = blockIdx.x * 64;
  const int wave = threadIdx.x >> 5;
  const int lane = threadIdx.x & 31;
  const int m0   = blockIdx.y * 128 + wave * 16;
  const int col  = lane & 15;
  const int hi   = lane >> 4;  // half-wave select

  const TX* Xb = X + (size_t)b * K * TT;

  // staging coords (per thread): 8 contiguous n for one k row
  const int kk = threadIdx.x >> 3;        // 0..31
  const int nn = (threadIdx.x & 7) * 8;   // 0..56 step 8

  auto stage = [&](int k0, int buf) {
    const TX* src = Xb + (size_t)(k0 + kk) * TT + n0 + nn;
#pragma unroll
    for (int j = 0; j < 8; ++j)
      lds_x[buf][(nn + j) * LDS_XK + kk] = (half_t)src[j];
    if (k0 + 64 < K)  // L2 prefetch of the tile after next
      __builtin_prefetch(Xb + (size_t)(k0 + 64 + kk) * TT + n0 + nn, 0, 1);
  };

  v8f zero = {};
  v8f acc[4];
#pragma unroll
  for (int j = 0; j < 4; ++j) acc[j] = zero;

  const int row = m0 + col;  // A-matrix row for this lane

  stage(0, 0);
  __syncthreads();

  for (int k0 = 0; k0 < K; k0 += 32) {
    const int buf = (k0 >> 5) & 1;
    if (k0 + 32 < K) stage(k0 + 32, buf ^ 1);

    // ---- A fragment from W (16x32 f16 A-layout) ----
    // lane<16: K = {0..7, 16..23}; lane>=16: K = {8..15, 24..31}
    FragH a;
    {
      const float* wp = W + (size_t)row * K + k0 + hi * 8;
#pragma unroll
      for (int i = 0; i < 8; ++i) a.v[i] = (half_t)wp[i];
#pragma unroll
      for (int i = 0; i < 8; ++i) a.v[8 + i] = (half_t)wp[16 + i];
    }

    // ---- 4 B fragments from LDS + WMMA ----
    // B-layout: element i <-> K row (hi*8 + i) for i<8, (16 + hi*8 + i-8) for i>=8, N col = lane&15
#pragma unroll
    for (int j = 0; j < 4; ++j) {
      const half_t* bp = &lds_x[buf][(j * 16 + col) * LDS_XK + hi * 8];
      FragH bb;
      bb.h[0] = *(const v8h*)bp;
      bb.h[1] = *(const v8h*)(bp + 16);
      acc[j] = __builtin_amdgcn_wmma_f32_16x16x32_f16(
          false, a.v, false, bb.v, (short)0, acc[j], false, false);
    }
    __syncthreads();
  }

  // ---- epilogue: C layout (lane holds col n, rows r + 8*hi) ----
#pragma unroll
  for (int j = 0; j < 4; ++j) {
    const int n = n0 + j * 16 + col;
#pragma unroll
    for (int r = 0; r < 8; ++r) {
      const int m = m0 + hi * 8 + r;
      Y[(size_t)b * M * TT + (size_t)m * TT + n] = acc[j][r] + bias[m];
    }
  }
}

// ---------------------------------------------------------------------------
// RoPE + pack to f16: src [B, H*64, T] fp32 -> dst [B, H, T, 64] f16
// one thread handles one rotation pair (i, i+32)
// ---------------------------------------------------------------------------
__global__ __launch_bounds__(256) void rope_pack_kernel(
    const float* __restrict__ src, half_t* __restrict__ dst, int H)
{
  size_t idx = (size_t)blockIdx.x * blockDim.x + threadIdx.x;
  size_t total = (size_t)BB * H * TT * 32;
  if (idx >= total) return;

  const int i = (int)(idx & 31);
  size_t r = idx >> 5;
  const int t = (int)(r % TT); r /= TT;
  const int h = (int)(r % H);
  const int b = (int)(r / H);

  const float* sp = src + ((size_t)(b * H + h) * 64) * TT + t;
  const float x1 = sp[(size_t)i * TT];
  const float x2 = sp[(size_t)(i + 32) * TT];

  const float inv = __powf(10000.0f, -(float)i / 32.0f);
  const float ang = (float)t * inv;
  float s, c;
  __sincosf(ang, &s, &c);

  half_t* dp = dst + ((size_t)(b * H + h) * TT + t) * 64;
  dp[i]      = (half_t)(x1 * c - x2 * s);
  dp[i + 32] = (half_t)(x2 * c + x1 * s);
}

// plain fp32 -> f16 copy (keeps V in d-major [B, h*64+d, T] layout)
__global__ __launch_bounds__(256) void f32_to_f16_kernel(
    const float* __restrict__ src, half_t* __restrict__ dst, size_t n)
{
  size_t i = (size_t)blockIdx.x * blockDim.x + threadIdx.x;
  if (i < n) dst[i] = (half_t)src[i];
}

// ---------------------------------------------------------------------------
// Flash attention: one wave per (b, h, 16-query tile); 64-key blocks.
//   qh [B, HQ, T, 64]  f16 (rope'd)
//   kh [B, HKV, T, 64] f16 (rope'd)
//   vt [B, HKV*64, T]  f16 (d-major -> contiguous B-frag loads for P*V)
//   oh [B, C, T]       f16 (attention output, ready for Wo GEMM)
// ---------------------------------------------------------------------------
__global__ __launch_bounds__(256) void flash_attn_wmma(
    const half_t* __restrict__ qh, const half_t* __restrict__ kh,
    const half_t* __restrict__ vt, half_t* __restrict__ oh, float scale)
{
  __shared__ half_t p_lds_all[8 * 16 * 80];  // per-wave 16x64 P slab, stride 80

  const int wave = threadIdx.x >> 5;
  const int lane = threadIdx.x & 31;
  const int col  = lane & 15;
  const int hi   = lane >> 4;
  half_t* p_lds = p_lds_all + wave * 16 * 80;

  const int wg = blockIdx.x * 8 + wave;   // 0 .. B*HQ*(T/16)-1 = 8191
  const int qt = wg & 127;                // query tile (T/16 = 128)
  const int h  = (wg >> 7) & 15;
  const int b  = wg >> 11;
  const int hk = h >> 2;                  // GQA: 4 query heads per kv head

  // ---- Q fragments (A layout), ks=0 dims 0-31, ks=1 dims 32-63 ----
  FragH qf[2];
  {
    const half_t* qp = qh + (((size_t)(b * HQ + h) * TT) + qt * 16 + col) * 64;
#pragma unroll
    for (int ks = 0; ks < 2; ++ks) {
      const int kb = ks * 32 + hi * 8;
      qf[ks].h[0] = *(const v8h*)(qp + kb);
      qf[ks].h[1] = *(const v8h*)(qp + kb + 16);
    }
  }

  v8f zero = {};
  float m_i[8], l_i[8];
  v8f o_acc[4];
#pragma unroll
  for (int r = 0; r < 8; ++r) { m_i[r] = -1e30f; l_i[r] = 0.0f; }
#pragma unroll
  for (int j = 0; j < 4; ++j) o_acc[j] = zero;

  const half_t* kbp = kh + ((size_t)(b * HKV + hk) * TT) * 64;
  const half_t* vbp = vt + ((size_t)(b * HKV + hk) * 64) * TT;

  for (int kb0 = 0; kb0 < TT; kb0 += 64) {
    // L2 prefetch for the next 64-key block (speculative, dropped at edges)
    if (kb0 + 64 < TT) {
      __builtin_prefetch(kbp + (size_t)(kb0 + 64 + lane * 2) * 64, 0, 1);
      __builtin_prefetch(vbp + (size_t)(lane * 2) * TT + kb0 + 64, 0, 1);
    }

    // ---- scores: 4 key subtiles of 16, 2 WMMAs each (d=64) ----
    v8f S[4];
#pragma unroll
    for (int j = 0; j < 4; ++j) {
      v8f sacc = zero;
      const int key = kb0 + j * 16 + col;
#pragma unroll
      for (int ks = 0; ks < 2; ++ks) {
        // B fragment of K^T: row = head dim, col = key position
        const half_t* kp = kbp + (size_t)key * 64 + ks * 32 + hi * 8;
        FragH kf;
        kf.h[0] = *(const v8h*)kp;
        kf.h[1] = *(const v8h*)(kp + 16);
        sacc = __builtin_amdgcn_wmma_f32_16x16x32_f16(
            false, qf[ks].v, false, kf.v, (short)0, sacc, false, false);
      }
      S[j] = sacc;
    }

    // ---- online softmax: row max over 64 keys (16-lane butterfly) ----
    float rm[8];
#pragma unroll
    for (int r = 0; r < 8; ++r)
      rm[r] = fmaxf(fmaxf(S[0][r] * scale, S[1][r] * scale),
                    fmaxf(S[2][r] * scale, S[3][r] * scale));
#pragma unroll
    for (int d = 1; d < 16; d <<= 1)
#pragma unroll
      for (int r = 0; r < 8; ++r)
        rm[r] = fmaxf(rm[r], __shfl_xor(rm[r], d, 32));

    float alpha[8], rs[8];
#pragma unroll
    for (int r = 0; r < 8; ++r) {
      const float mnew = fmaxf(m_i[r], rm[r]);
      alpha[r] = __expf(m_i[r] - mnew);
      m_i[r] = mnew;
      rs[r] = 0.0f;
    }

    // ---- P = exp(S*scale - m); spill to LDS to rotate C-layout -> A-layout
#pragma unroll
    for (int j = 0; j < 4; ++j)
#pragma unroll
      for (int r = 0; r < 8; ++r) {
        const float p = __expf(S[j][r] * scale - m_i[r]);
        rs[r] += p;
        p_lds[(hi * 8 + r) * 80 + j * 16 + col] = (half_t)p;
      }

#pragma unroll
    for (int d = 1; d < 16; d <<= 1)
#pragma unroll
      for (int r = 0; r < 8; ++r)
        rs[r] += __shfl_xor(rs[r], d, 32);

#pragma unroll
    for (int r = 0; r < 8; ++r)
      l_i[r] = l_i[r] * alpha[r] + rs[r];
#pragma unroll
    for (int j = 0; j < 4; ++j)
#pragma unroll
      for (int r = 0; r < 8; ++r)
        o_acc[j][r] *= alpha[r];

    // ---- O += P * V : P as A frags (from LDS, DS ops in-order per wave),
    //      V as B frags (contiguous d-major loads)
#pragma unroll
    for (int ks = 0; ks < 2; ++ks) {
      FragH pf;
      {
        const half_t* pp = p_lds + col * 80 + ks * 32 + hi * 8;
        pf.h[0] = *(const v8h*)pp;
        pf.h[1] = *(const v8h*)(pp + 16);
      }
#pragma unroll
      for (int jd = 0; jd < 4; ++jd) {
        const int dcol = jd * 16 + col;
        const half_t* vp = vbp + (size_t)dcol * TT + kb0 + ks * 32 + hi * 8;
        FragH vf;
        vf.h[0] = *(const v8h*)vp;
        vf.h[1] = *(const v8h*)(vp + 16);
        o_acc[jd] = __builtin_amdgcn_wmma_f32_16x16x32_f16(
            false, pf.v, false, vf.v, (short)0, o_acc[jd], false, false);
      }
    }
  }

  // ---- epilogue: normalize, pack 8 t-consecutive halfs, b128 store ----
#pragma unroll
  for (int jd = 0; jd < 4; ++jd) {
    const int dcol = jd * 16 + col;
    half_t* op = oh + ((size_t)b * CC + h * 64 + dcol) * TT + qt * 16 + hi * 8;
    v8h ov;
#pragma unroll
    for (int r = 0; r < 8; ++r)
      ov[r] = (half_t)(o_acc[jd][r] / l_i[r]);
    *(v8h*)op = ov;
  }
}

// ---------------------------------------------------------------------------
extern "C" void kernel_launch(void* const* d_in, const int* in_sizes, int n_in,
                              void* d_out, int out_size, void* d_ws, size_t ws_size,
                              hipStream_t stream) {
  (void)in_sizes; (void)n_in; (void)out_size; (void)ws_size;

  const float* query = (const float*)d_in[0];  // [B, C, T]
  const float* keyv  = (const float*)d_in[1];  // [B, C, T]
  const float* Wq    = (const float*)d_in[2];  // [C, C]
  const float* bq    = (const float*)d_in[3];
  const float* Wk    = (const float*)d_in[4];  // [KV, C]
  const float* bk    = (const float*)d_in[5];
  const float* Wv    = (const float*)d_in[6];  // [KV, C]
  const float* bv    = (const float*)d_in[7];
  const float* Wo    = (const float*)d_in[8];  // [C, C]
  const float* bo    = (const float*)d_in[9];
  float* out = (float*)d_out;                  // [B, C, T]

  // workspace carve-up
  char* ws = (char*)d_ws;
  size_t off = 0;
  float*  q32 = (float*)(ws + off);  off += (size_t)BB * CC * TT * 4;     // 32 MB
  float*  k32 = (float*)(ws + off);  off += (size_t)BB * KVDIM * TT * 4;  //  8 MB
  float*  v32 = (float*)(ws + off);  off += (size_t)BB * KVDIM * TT * 4;  //  8 MB
  half_t* qhp = (half_t*)(ws + off); off += (size_t)BB * CC * TT * 2;     // 16 MB
  half_t* khp = (half_t*)(ws + off); off += (size_t)BB * KVDIM * TT * 2;  //  4 MB
  half_t* vhp = (half_t*)(ws + off); off += (size_t)BB * KVDIM * TT * 2;  //  4 MB
  half_t* ohp = (half_t*)(ws + off); off += (size_t)BB * CC * TT * 2;     // 16 MB

  const dim3 blk(256);

  // 1-3: input projections (fp32 inputs, fp32 outputs)
  gemm_bias_wmma<float><<<dim3(TT / 64, CC / 128, BB), blk, 0, stream>>>(
      Wq, query, bq, q32, CC, CC);
  gemm_bias_wmma<float><<<dim3(TT / 64, KVDIM / 128, BB), blk, 0, stream>>>(
      Wk, keyv, bk, k32, KVDIM, CC);
  gemm_bias_wmma<float><<<dim3(TT / 64, KVDIM / 128, BB), blk, 0, stream>>>(
      Wv, keyv, bv, v32, KVDIM, CC);

  // 4-5: RoPE + f16 pack into [B, H, T, 64]
  {
    const size_t nq = (size_t)BB * HQ * TT * 32;
    rope_pack_kernel<<<dim3((unsigned)((nq + 255) / 256)), blk, 0, stream>>>(
        q32, qhp, HQ);
    const size_t nk = (size_t)BB * HKV * TT * 32;
    rope_pack_kernel<<<dim3((unsigned)((nk + 255) / 256)), blk, 0, stream>>>(
        k32, khp, HKV);
  }

  // 6: V -> f16 (stays d-major [B, h*64+d, T])
  {
    const size_t nv = (size_t)BB * KVDIM * TT;
    f32_to_f16_kernel<<<dim3((unsigned)((nv + 255) / 256)), blk, 0, stream>>>(
        v32, vhp, nv);
  }

  // 7: flash attention (8 waves/block, one wave per (b,h,qtile))
  {
    const int total_waves = BB * HQ * (TT / 16);  // 8192
    flash_attn_wmma<<<dim3(total_waves / 8), blk, 0, stream>>>(
        qhp, khp, vhp, ohp, 0.125f /* 1/sqrt(64) */);
  }

  // 8: output projection (f16 input, fp32 output)
  gemm_bias_wmma<_Float16><<<dim3(TT / 64, CC / 128, BB), blk, 0, stream>>>(
      Wo, ohp, bo, out, CC, CC);
}
